// QuantLinear_21861383537485
// MI455X (gfx1250) — matrix operand order repaired and checked by reading, not compiled, not run on hardware
//
#include <hip/hip_runtime.h>
#include <hip/hip_bf16.h>

typedef __attribute__((ext_vector_type(16))) _Float16     v16h;
typedef __attribute__((ext_vector_type(8)))  _Float16     v8h;
typedef __attribute__((ext_vector_type(2)))  _Float16     v2h;
typedef __attribute__((ext_vector_type(8)))  float        v8f;
typedef __attribute__((ext_vector_type(4)))  int          v4i;

#define K_DIM 4096
#define N_DIM 11008
#define BM 128
#define BN 128
#define BK 32
#define LDSS 40            // 32 halves + 8 pad (80 B row stride, 16B-aligned)
#define KT (K_DIM / BK)    // 128 k-steps

#if __has_builtin(__builtin_amdgcn_global_load_async_to_lds_b128)
#define HAS_ASYNC_LDS 1
typedef __attribute__((address_space(1))) v4i* gv4i_p;   // global int4*
typedef __attribute__((address_space(3))) v4i* lv4i_p;   // LDS int4*
#else
#define HAS_ASYNC_LDS 0
#endif

// ---- packed int8 -> fp16 conversion (exact) ----
// perm: {0x00,bHi,0x00,bLo}; xor 0x64806480: each half = fp16(1024 + (b+128));
// pk_add -1152: each half = fp16(b).  3 VALU ops per 2 elements.
__device__ __forceinline__ unsigned cvt_i8pair_f16(unsigned d, unsigned sel) {
    unsigned p = __builtin_amdgcn_perm(0u, d, sel);   // v_perm_b32
    p ^= 0x64806480u;                                  // v_xor_b32
    v2h h = __builtin_bit_cast(v2h, p);
    h += (v2h){(_Float16)-1152.f, (_Float16)-1152.f};  // v_pk_add_f16
    return __builtin_bit_cast(unsigned, h);
}

__device__ __forceinline__ void stageB_convert(uint4 d, _Float16* dst) {
    uint4 lo, hi;
    lo.x = cvt_i8pair_f16(d.x, 0x0C010C00u);
    lo.y = cvt_i8pair_f16(d.x, 0x0C030C02u);
    lo.z = cvt_i8pair_f16(d.y, 0x0C010C00u);
    lo.w = cvt_i8pair_f16(d.y, 0x0C030C02u);
    hi.x = cvt_i8pair_f16(d.z, 0x0C010C00u);
    hi.y = cvt_i8pair_f16(d.z, 0x0C030C02u);
    hi.z = cvt_i8pair_f16(d.w, 0x0C010C00u);
    hi.w = cvt_i8pair_f16(d.w, 0x0C030C02u);
    *(uint4*)dst       = lo;   // ds_store_b128
    *(uint4*)(dst + 8) = hi;   // ds_store_b128
}

__device__ __forceinline__ void stageA(const _Float16* g, _Float16* lds) {
#if HAS_ASYNC_LDS
    __builtin_amdgcn_global_load_async_to_lds_b128(
        (gv4i_p)(void*)g, (lv4i_p)(void*)lds, 0, 0);
#else
    *(uint4*)lds = *(const uint4*)g;
#endif
}

__device__ __forceinline__ void drain_async() {
#if HAS_ASYNC_LDS
#if __has_builtin(__builtin_amdgcn_s_wait_asynccnt)
    __builtin_amdgcn_s_wait_asynccnt(0);
#else
    asm volatile("s_wait_asynccnt 0x0" ::: "memory");
#endif
#endif
}

__global__ __launch_bounds__(256)
void qlin_wmma_f16(const _Float16* __restrict__ X,
                   const signed char* __restrict__ Q,
                   const _Float16* __restrict__ S,
                   _Float16* __restrict__ O)
{
    __shared__ _Float16 As[2][BM * LDSS];   // activations, fp16
    __shared__ _Float16 Bs[2][BN * LDSS];   // weights dequantized to fp16 ints

    const int tid   = threadIdx.x;
    const int lane  = tid & 31;
    const int wave  = tid >> 5;       // 8 waves
    const int waveM = wave & 1;       // 2 along M (64 rows each)
    const int waveN = wave >> 1;      // 4 along N (32 cols each)
    const int r16   = lane & 15;
    const int hi    = lane >> 4;      // lane half selects K sub-chunk

    const long mBlock = (long)blockIdx.y * BM;
    const long nBlock = (long)blockIdx.x * BN;

    // A tile: 128 rows x 64 B -> 512 x 16B chunks, 2 per thread
    const int aRow = tid >> 2;             // + 64 for second chunk
    const int aCol = (tid & 3) * 8;        // halves
    // B tile: 128 rows x 32 B int8 -> 256 x 16B chunks, 1 per thread
    const int bRow = tid >> 1;
    const int bCol = (tid & 1) * 16;

    const _Float16*    xA = X + (mBlock + aRow) * (long)K_DIM + aCol;
    const _Float16*    xB = X + (mBlock + aRow + 64) * (long)K_DIM + aCol;
    const signed char* qP = Q + (nBlock + bRow) * (long)K_DIM + bCol;

    v8f acc[4][2];
#pragma unroll
    for (int mf = 0; mf < 4; ++mf)
#pragma unroll
        for (int nf = 0; nf < 2; ++nf) acc[mf][nf] = (v8f){};

    // ---- prologue: stage tile 0 into buffer 0 ----
    {
        stageA(xA, &As[0][aRow * LDSS + aCol]);
        stageA(xB, &As[0][(aRow + 64) * LDSS + aCol]);
        uint4 q0 = *(const uint4*)qP;
        stageB_convert(q0, &Bs[0][bRow * LDSS + bCol]);
        drain_async();
    }
    __syncthreads();

    for (int kt = 0; kt < KT; ++kt) {
        const int cur = kt & 1;
        const int nxt = cur ^ 1;
        const bool pf = (kt + 1) < KT;

        // issue next-tile traffic first: latency hides under WMMA
        uint4 qreg;
        if (pf) {
            const long ko = (long)(kt + 1) * BK;
            stageA(xA + ko, &As[nxt][aRow * LDSS + aCol]);
            stageA(xB + ko, &As[nxt][(aRow + 64) * LDSS + aCol]);
            qreg = *(const uint4*)(qP + ko);
        }

        // ---- compute: 4x2 fragments, one K=32 WMMA each ----
        v16h afr[4];
#pragma unroll
        for (int mf = 0; mf < 4; ++mf) {
            const int m0 = waveM * 64 + mf * 16 + r16;
            v8h alo = *(const v8h*)&As[cur][m0 * LDSS + hi * 16];       // K 0-7 / 16-23
            v8h ahi = *(const v8h*)&As[cur][m0 * LDSS + hi * 16 + 8];   // K 8-15 / 24-31
            afr[mf] = __builtin_shufflevector(alo, ahi,
                        0,1,2,3,4,5,6,7,8,9,10,11,12,13,14,15);
        }
#pragma unroll
        for (int nf = 0; nf < 2; ++nf) {
            const int n0 = waveN * 32 + nf * 16 + r16;
            v16h bfr = *(const v16h*)&Bs[cur][n0 * LDSS + hi * 16];
#pragma unroll
            for (int mf = 0; mf < 4; ++mf) {
                acc[mf][nf] = __builtin_amdgcn_wmma_f32_16x16x32_f16(
                    false, afr[mf], false, bfr,
                    (short)0, acc[mf][nf], false, false);
            }
        }

        // ---- publish next tile ----
        if (pf) {
            stageB_convert(qreg, &Bs[nxt][bRow * LDSS + bCol]);
            drain_async();
        }
        __syncthreads();
    }

    // ---- epilogue: per-output-channel scale in f32, store fp16 ----
#pragma unroll
    for (int nf = 0; nf < 2; ++nf) {
        const long col = nBlock + waveN * 32 + nf * 16 + r16;
        const float s = (float)S[col];
#pragma unroll
        for (int mf = 0; mf < 4; ++mf) {
            const long rowBase = mBlock + waveM * 64 + mf * 16 + hi * 8;
#pragma unroll
            for (int r = 0; r < 8; ++r) {
                O[(rowBase + r) * (long)N_DIM + col] =
                    (_Float16)(acc[mf][nf][r] * s);
            }
        }
    }
}

extern "C" void kernel_launch(void* const* d_in, const int* in_sizes, int n_in,
                              void* d_out, int out_size, void* d_ws, size_t ws_size,
                              hipStream_t stream) {
    const _Float16*    X = (const _Float16*)d_in[0];     // [8192, 4096] fp16
    const signed char* Q = (const signed char*)d_in[1];  // [11008, 4096] int8
    const _Float16*    S = (const _Float16*)d_in[2];     // [11008, 1] fp16
    _Float16*          O = (_Float16*)d_out;             // [8192, 11008] fp16

    const int M = in_sizes[0] / K_DIM;                   // 8192
    dim3 grid(N_DIM / BN, M / BM);                       // (86, 64)
    qlin_wmma_f16<<<grid, 256, 0, stream>>>(X, Q, S, O);
}
